// LoRAAttention_15204184228289
// MI455X (gfx1250) — matrix operand order repaired and checked by reading, not compiled, not run on hardware
//
#include <hip/hip_runtime.h>
#include <hip/hip_bf16.h>

#define D_MODEL 768
#define N_HEADS 12
#define D_K 64
#define SEQ 4096
#define RANK 8
#define LORA_SCALING 2.0f      // alpha/rank = 16/8
#define QK_SCALE 0.125f        // 1/sqrt(64)
#define LOG2E 1.44269504088896340736f

typedef __attribute__((ext_vector_type(16))) __bf16 v16bf;
typedef __attribute__((ext_vector_type(8)))  __bf16 v8bf;
typedef __attribute__((ext_vector_type(8)))  float  v8f;

static __device__ __forceinline__ v16bf make_frag(const __bf16* lo, const __bf16* hi) {
    v8bf a = *(const v8bf*)lo;
    v8bf b = *(const v8bf*)hi;
    v16bf r;
#pragma unroll
    for (int i = 0; i < 8; ++i) { r[i] = a[i]; r[8 + i] = b[i]; }
    return r;
}

static __device__ __forceinline__ v8f wmma_bf16(v16bf a, v16bf b, v8f c) {
    return __builtin_amdgcn_wmma_f32_16x16x32_bf16(
        /*neg_a=*/false, a, /*neg_b=*/false, b,
        /*c_mod=*/(short)0, c, /*reuse_a=*/false, /*reuse_b=*/false);
}

// Async 16-byte global -> LDS copy (CDNA5 GLOBAL_LOAD_ASYNC_TO_LDS_B128,
// tracked by ASYNCcnt). LDS byte offset is the low 32 bits of the generic
// pointer to a __shared__ object.
static __device__ __forceinline__ void async_copy16(const __bf16* gsrc, __bf16* lds_dst) {
    unsigned int lds_addr = (unsigned int)(unsigned long long)lds_dst;
    unsigned long long gaddr = (unsigned long long)gsrc;
    asm volatile("global_load_async_to_lds_b128 %0, %1, off"
                 :: "v"(lds_addr), "v"(gaddr)
                 : "memory");
}

// ---------------------------------------------------------------------------
// Convert fp32 activations -> bf16
// ---------------------------------------------------------------------------
__global__ __launch_bounds__(256)
void convert_x_kernel(const float* __restrict__ x, __bf16* __restrict__ xb) {
    int i = blockIdx.x * blockDim.x + threadIdx.x;
    xb[i] = (__bf16)x[i];
}

// ---------------------------------------------------------------------------
// W_eff = W + 2 * Wup @ Wdown   (rank-8 LoRA fold), stored TRANSPOSED (K-major)
// ---------------------------------------------------------------------------
__global__ __launch_bounds__(256)
void fold_lora_kernel(const float* __restrict__ w,
                      const float* __restrict__ wd,   // [RANK][D_MODEL] or null
                      const float* __restrict__ wu,   // [D_MODEL][RANK] or null
                      __bf16* __restrict__ wt) {
    int idx = blockIdx.x * blockDim.x + threadIdx.x;   // over 768*768
    int e = idx / D_MODEL;
    int d = idx % D_MODEL;
    float acc = w[(size_t)e * D_MODEL + d];
    if (wd != nullptr) {
        float s = 0.f;
#pragma unroll
        for (int r = 0; r < RANK; ++r)
            s += wu[e * RANK + r] * wd[r * D_MODEL + d];
        acc += LORA_SCALING * s;
    }
    wt[(size_t)d * D_MODEL + e] = (__bf16)acc;
}

// ---------------------------------------------------------------------------
// GEMM: C[SEQ][768] = A_bf16[SEQ][768] x Wt_bf16[768][768](K-major) + bias
// One wave computes a 16x64 output tile; block = 4 waves (64x64 tile).
// ---------------------------------------------------------------------------
__global__ __launch_bounds__(128)
void gemm_kernel(const __bf16* __restrict__ A,
                 const __bf16* __restrict__ Wt,
                 const float*  __restrict__ bias,
                 __bf16* __restrict__ outb,
                 float*  __restrict__ outf,
                 float scale, int transposeOut) {
    int wave    = threadIdx.x >> 5;
    int lane    = threadIdx.x & 31;
    int lrow    = lane & 15;
    int halfSel = lane >> 4;

    int mbase = blockIdx.y * 64 + wave * 16;
    int nbase = blockIdx.x * 64;

    v8f acc[4] = {};

    const __bf16* Abase = A + (size_t)(mbase + lrow) * D_MODEL;
    int koff = halfSel * 8;

    for (int k0 = 0; k0 < D_MODEL; k0 += 32) {
        v16bf af = make_frag(Abase + k0 + koff, Abase + k0 + koff + 16);
        const __bf16* Brow = Wt + (size_t)(k0 + koff) * D_MODEL + nbase + lrow;
#pragma unroll
        for (int t = 0; t < 4; ++t) {
            v16bf bfrag;
#pragma unroll
            for (int i = 0; i < 8; ++i) {
                bfrag[i]     = Brow[(size_t)i * D_MODEL + t * 16];
                bfrag[8 + i] = Brow[(size_t)(16 + i) * D_MODEL + t * 16];
            }
            acc[t] = wmma_bf16(af, bfrag, acc[t]);
        }
    }

#pragma unroll
    for (int t = 0; t < 4; ++t) {
        int col = nbase + t * 16 + lrow;
        float bv = bias ? bias[col] : 0.0f;
#pragma unroll
        for (int v = 0; v < 8; ++v) {
            int row = mbase + halfSel * 8 + v;
            float val = (acc[t][v] + bv) * scale;
            if (outf)               outf[(size_t)row * D_MODEL + col] = val;
            else if (transposeOut)  outb[(size_t)col * SEQ + row]     = (__bf16)val;
            else                    outb[(size_t)row * D_MODEL + col] = (__bf16)val;
        }
    }
}

// ---------------------------------------------------------------------------
// Fused flash attention with async global->LDS staging of the K/V chunk
// shared by all 8 waves of the block (double-buffered).
// Q[s][768] bf16 (pre-scaled 1/8), Kt[768][4096] bf16, V[s][768] bf16.
// ---------------------------------------------------------------------------
__global__ __launch_bounds__(256)
void attention_kernel(const __bf16* __restrict__ Q,
                      const __bf16* __restrict__ Kt,
                      const __bf16* __restrict__ V,
                      __bf16* __restrict__ O) {
    __shared__ __align__(16) __bf16 ktile[2][64][40];   // [buf][d][key], padded rows
    __shared__ __align__(16) __bf16 vtile[2][32][72];   // [buf][key][d], padded rows
    __shared__ __align__(16) __bf16 pbuf[8][16][40];    // per-wave P tile

    int tid     = threadIdx.x;
    int wave    = tid >> 5;
    int lane    = tid & 31;
    int lrow    = lane & 15;
    int halfSel = lane >> 4;
    int koff    = halfSel * 8;

    int head  = blockIdx.y;
    int qbase = blockIdx.x * 128 + wave * 16;
    int hcol  = head * D_K;

    // Per-thread async-copy routing: one 16B chunk of the K tile and one of
    // the V tile per thread per 32-key step (256 threads x 16B = 4KB each).
    int kd = tid >> 2, kp = tid & 3;   // K tile: d row 0..63, chunk 0..3
    int vr = tid >> 3, vp = tid & 7;   // V tile: key row 0..31, chunk 0..7
    const __bf16* kg = Kt + (size_t)(hcol + kd) * SEQ + kp * 8;
    const __bf16* vg = V + (size_t)vr * D_MODEL + hcol + vp * 8;

    // Persistent Q A-fragments (d = 0..31 and 32..63)
    const __bf16* qrow = Q + (size_t)(qbase + lrow) * D_MODEL + hcol;
    v16bf qa0 = make_frag(qrow + koff,      qrow + koff + 16);
    v16bf qa1 = make_frag(qrow + 32 + koff, qrow + 32 + koff + 16);

    v8f  o[4] = {};
    float rmax[8], rsum[8];
#pragma unroll
    for (int v = 0; v < 8; ++v) { rmax[v] = -3.0e38f; rsum[v] = 0.0f; }

    // Prime buffer 0 with the first chunk.
    async_copy16(kg, &ktile[0][kd][kp * 8]);
    async_copy16(vg, &vtile[0][vr][vp * 8]);

    int cur = 0;
    for (int kb = 0; kb < SEQ; kb += 32) {
        // Own async copies for buf[cur] complete, then cross-wave handoff.
        asm volatile("s_wait_asynccnt 0x0" ::: "memory");
        __syncthreads();

        // Prefetch next chunk into the other buffer (its previous readers all
        // passed the barrier above).
        int kn = kb + 32;
        if (kn < SEQ) {
            async_copy16(kg + kn, &ktile[cur ^ 1][kd][kp * 8]);
            async_copy16(vg + (size_t)kn * D_MODEL, &vtile[cur ^ 1][vr][vp * 8]);
        }

        // scores: two 16x16 key tiles, each = Q(16x64) x K^T(64x16) in 2 wmma
        v8f s[2];
#pragma unroll
        for (int t = 0; t < 2; ++t) {
            v16bf b0, b1;
#pragma unroll
            for (int i = 0; i < 8; ++i) {
                b0[i]     = ktile[cur][koff + i][t * 16 + lrow];
                b0[8 + i] = ktile[cur][koff + 16 + i][t * 16 + lrow];
                b1[i]     = ktile[cur][32 + koff + i][t * 16 + lrow];
                b1[8 + i] = ktile[cur][32 + koff + 16 + i][t * 16 + lrow];
            }
            v8f c = {};
            c = wmma_bf16(qa0, b0, c);
            c = wmma_bf16(qa1, b1, c);
            s[t] = c;
        }

        // online softmax (row = VGPR v within a 16-lane half; cols across lanes)
        float rescale[8];
#pragma unroll
        for (int v = 0; v < 8; ++v) {
            float tmax = fmaxf(s[0][v], s[1][v]);
#pragma unroll
            for (int off = 8; off >= 1; off >>= 1)
                tmax = fmaxf(tmax, __shfl_xor(tmax, off, 32));
            float newm = fmaxf(rmax[v], tmax);
            float rs = exp2f((rmax[v] - newm) * LOG2E);
            rmax[v] = newm;
            float p0 = exp2f((s[0][v] - newm) * LOG2E);
            float p1 = exp2f((s[1][v] - newm) * LOG2E);
            float ps = p0 + p1;
#pragma unroll
            for (int off = 8; off >= 1; off >>= 1)
                ps += __shfl_xor(ps, off, 32);
            rsum[v] = rsum[v] * rs + ps;
            rescale[v] = rs;
            int m = v + 8 * halfSel;
            pbuf[wave][m][lrow]      = (__bf16)p0;
            pbuf[wave][m][lrow + 16] = (__bf16)p1;
        }
#pragma unroll
        for (int f = 0; f < 4; ++f)
#pragma unroll
            for (int v = 0; v < 8; ++v) o[f][v] *= rescale[v];

        asm volatile("s_wait_dscnt 0x0" ::: "memory");   // P visible to same-wave reads

        // P as A-fragment (16x32) from per-wave LDS
        v16bf pa;
#pragma unroll
        for (int i = 0; i < 8; ++i) {
            pa[i]     = pbuf[wave][lrow][koff + i];
            pa[8 + i] = pbuf[wave][lrow][koff + 16 + i];
        }

        // O += P(16x32) x V(32x64); V B-frags from the shared LDS tile
#pragma unroll
        for (int f = 0; f < 4; ++f) {
            v16bf vb;
#pragma unroll
            for (int i = 0; i < 8; ++i) {
                vb[i]     = vtile[cur][koff + i][f * 16 + lrow];
                vb[8 + i] = vtile[cur][koff + 16 + i][f * 16 + lrow];
            }
            o[f] = wmma_bf16(pa, vb, o[f]);
        }

        cur ^= 1;
    }

    // normalize and store
#pragma unroll
    for (int f = 0; f < 4; ++f) {
        int col = hcol + f * 16 + lrow;
#pragma unroll
        for (int v = 0; v < 8; ++v) {
            int row = qbase + halfSel * 8 + v;
            O[(size_t)row * D_MODEL + col] = (__bf16)(o[f][v] / rsum[v]);
        }
    }
}

// ---------------------------------------------------------------------------
// Host launch
// ---------------------------------------------------------------------------
extern "C" void kernel_launch(void* const* d_in, const int* in_sizes, int n_in,
                              void* d_out, int out_size, void* d_ws, size_t ws_size,
                              hipStream_t stream) {
    const float* x       = (const float*)d_in[0];
    const float* wq      = (const float*)d_in[1];
    const float* bq      = (const float*)d_in[2];
    const float* wk      = (const float*)d_in[3];
    const float* bk      = (const float*)d_in[4];
    const float* wv      = (const float*)d_in[5];
    const float* bv      = (const float*)d_in[6];
    const float* wo      = (const float*)d_in[7];
    const float* bo      = (const float*)d_in[8];
    const float* wq_down = (const float*)d_in[9];
    const float* wq_up   = (const float*)d_in[10];
    const float* wk_down = (const float*)d_in[11];
    const float* wk_up   = (const float*)d_in[12];
    const float* wv_down = (const float*)d_in[13];
    const float* wv_up   = (const float*)d_in[14];
    float* out = (float*)d_out;

    char* ws = (char*)d_ws;
    size_t off = 0;
    const size_t ACT = (size_t)SEQ * D_MODEL * sizeof(__bf16);
    const size_t WMT = (size_t)D_MODEL * D_MODEL * sizeof(__bf16);
    __bf16* xb    = (__bf16*)(ws + off); off += ACT;
    __bf16* wtq   = (__bf16*)(ws + off); off += WMT;
    __bf16* wtk   = (__bf16*)(ws + off); off += WMT;
    __bf16* wtv   = (__bf16*)(ws + off); off += WMT;
    __bf16* wto   = (__bf16*)(ws + off); off += WMT;
    __bf16* qb    = (__bf16*)(ws + off); off += ACT;
    __bf16* ktb   = (__bf16*)(ws + off); off += ACT;   // [768][4096]
    __bf16* vb    = (__bf16*)(ws + off); off += ACT;
    __bf16* attnb = (__bf16*)(ws + off); off += ACT;

    convert_x_kernel<<<(SEQ * D_MODEL) / 256, 256, 0, stream>>>(x, xb);

    int wgrid = (D_MODEL * D_MODEL) / 256;
    fold_lora_kernel<<<wgrid, 256, 0, stream>>>(wq, wq_down, wq_up, wtq);
    fold_lora_kernel<<<wgrid, 256, 0, stream>>>(wk, wk_down, wk_up, wtk);
    fold_lora_kernel<<<wgrid, 256, 0, stream>>>(wv, wv_down, wv_up, wtv);
    fold_lora_kernel<<<wgrid, 256, 0, stream>>>(wo, nullptr, nullptr, wto);

    dim3 ggrid(D_MODEL / 64, SEQ / 64);
    gemm_kernel<<<ggrid, 128, 0, stream>>>(xb, wtq, bq, qb,  nullptr, QK_SCALE, 0);
    gemm_kernel<<<ggrid, 128, 0, stream>>>(xb, wtk, bk, ktb, nullptr, 1.0f,     1);
    gemm_kernel<<<ggrid, 128, 0, stream>>>(xb, wtv, bv, vb,  nullptr, 1.0f,     0);

    attention_kernel<<<dim3(SEQ / 128, N_HEADS), 256, 0, stream>>>(qb, ktb, vb, attnb);

    gemm_kernel<<<ggrid, 128, 0, stream>>>(attnb, wto, bo, nullptr, out, 1.0f, 0);
}
